// MultiHeadAttention_4363686773423
// MI455X (gfx1250) — matrix operand order repaired and checked by reading, not compiled, not run on hardware
//
#include <hip/hip_runtime.h>
#include <hip/hip_bf16.h>
#include <math.h>

// Problem constants (fixed by reference)
#define BB     4
#define LQn    2048
#define LKn    2048
#define DDn    1024
#define HHn    16
#define DKn    64
#define DVn    64
#define HDn    (HHn * DVn)      // 1024
#define INV_TEMP 0.125f         // 1/sqrt(64)
#define LN_EPS 1e-5f

typedef __bf16 bf16_t;
typedef __attribute__((ext_vector_type(16))) __bf16 v16bf;
typedef __attribute__((ext_vector_type(8)))  float  v8f;

union FragU { v16bf v; uint4 u[2]; };

// A-matrix fragment (16x32 bf16, M x K), source row-major [M, ld].
// Per ISA 7.12.2: lanes 0-15 -> M=lane, K=0..7 & 16..23 ; lanes 16-31 -> K=8..15 & 24..31.
__device__ __forceinline__ v16bf load_a16(const bf16_t* base, int lane, int ld) {
    const int m    = lane & 15;
    const int koff = (lane < 16) ? 0 : 8;
    const bf16_t* p = base + (size_t)m * ld + koff;
    FragU f;
    f.u[0] = *reinterpret_cast<const uint4*>(p);        // K = koff .. koff+7
    f.u[1] = *reinterpret_cast<const uint4*>(p + 16);   // K = koff+16 .. koff+23
    return f.v;
}

// B-matrix fragment (32x16 bf16, K x N), source stored K-contiguous per column:
// ptr layout [N, ld] with K contiguous. Lanes 0-15: N=lane, K=0..7 & 16..23; lanes 16-31: K=8..15 & 24..31.
__device__ __forceinline__ v16bf load_b16(const bf16_t* base, int lane, int ld) {
    const int n    = lane & 15;
    const int koff = (lane < 16) ? 0 : 8;
    const bf16_t* p = base + (size_t)n * ld + koff;
    FragU f;
    f.u[0] = *reinterpret_cast<const uint4*>(p);
    f.u[1] = *reinterpret_cast<const uint4*>(p + 16);
    return f.v;
}

__device__ __forceinline__ v8f wmma_bf16(v16bf a, v16bf b, v8f c) {
    return __builtin_amdgcn_wmma_f32_16x16x32_bf16(false, a, false, b, (short)0, c, false, false);
}

// ---------------------------------------------------------------------------
// fp32 -> bf16 elementwise convert
__global__ void cvt_bf16(const float* __restrict__ src, bf16_t* __restrict__ dst, int n) {
    int i = blockIdx.x * blockDim.x + threadIdx.x;
    const int stride = gridDim.x * blockDim.x;
    for (; i < n; i += stride) dst[i] = (bf16_t)src[i];
}

// Transpose-pack W [K, N] fp32 -> WT [N, K] bf16 (K contiguous per output row)
__global__ void pack_wT(const float* __restrict__ W, bf16_t* __restrict__ WT, int K, int N) {
    int i = blockIdx.x * blockDim.x + threadIdx.x;
    const int stride = gridDim.x * blockDim.x;
    const int total = K * N;
    for (; i < total; i += stride) {
        int n = i / K, k = i - n * K;
        WT[i] = (bf16_t)W[(size_t)k * N + n];
    }
}

// ---------------------------------------------------------------------------
// WMMA GEMM: X[M,K](bf16, row-major) x WT[N,K](bf16) + bias.
// Block = 256 threads (8 waves); each wave: 32(M) x 64(N); block tile 256 x 64.
// Two A fragments share each B fragment -> 8 WMMAs per 12 b128 loads per K-step.
// MODE 0: out bf16 head-major [B,H,L,64], value scaled by `scale` (Q/K proj)
// MODE 1: out bf16 transposed  [B,H,64,L]                         (V proj)
// MODE 2: out fp32 flat [M,N], += resid                           (out proj)
template <int MODE>
__global__ __launch_bounds__(256) void gemm_bf16(const bf16_t* __restrict__ X,
                                                 const bf16_t* __restrict__ WT,
                                                 const float* __restrict__ bias,
                                                 const float* __restrict__ resid,
                                                 void* __restrict__ outp,
                                                 int M, int K, int N, float scale) {
    const int lane = threadIdx.x & 31;
    const int w    = threadIdx.x >> 5;
    const int m0   = blockIdx.x * 256 + w * 32;
    const int n0   = blockIdx.y * 64;

    v8f acc[2][4] = {};
    for (int k = 0; k < K; k += 32) {
        const bf16_t* xa = X + (size_t)m0 * K + k;
        if (k + 32 < K) {                       // head start on next A slice
            __builtin_prefetch(xa + 32, 0, 0);
            __builtin_prefetch(xa + (size_t)16 * K + 32, 0, 0);
        }
        v16bf a0 = load_a16(xa, lane, K);
        v16bf a1 = load_a16(xa + (size_t)16 * K, lane, K);
#pragma unroll
        for (int nt = 0; nt < 4; nt++) {
            v16bf b = load_b16(WT + (size_t)(n0 + nt * 16) * K + k, lane, K);
            acc[0][nt] = wmma_bf16(a0, b, acc[0][nt]);
            acc[1][nt] = wmma_bf16(a1, b, acc[1][nt]);
        }
    }

    const int cl = lane & 15, half = lane >> 4;
#pragma unroll
    for (int mt = 0; mt < 2; mt++) {
#pragma unroll
        for (int nt = 0; nt < 4; nt++) {
            const int col = n0 + nt * 16 + cl;
            const float bv = bias[col];
#pragma unroll
            for (int r = 0; r < 8; r++) {
                const int row = m0 + mt * 16 + r + half * 8;
                float val = acc[mt][nt][r] + bv;
                if (MODE == 0 || MODE == 1) {
                    const int b = row >> 11;          // / 2048
                    const int l = row & 2047;
                    const int h = col >> 6;
                    const int d = col & 63;
                    bf16_t* out = (bf16_t*)outp;
                    if (MODE == 0)
                        out[(((size_t)b * HHn + h) * LQn + l) * 64 + d] = (bf16_t)(val * scale);
                    else
                        out[(((size_t)b * HHn + h) * 64 + d) * LKn + l] = (bf16_t)val;
                } else {
                    float* out = (float*)outp;
                    const size_t idx = (size_t)row * N + col;
                    out[idx] = val + resid[idx];
                }
            }
        }
    }
}

// ---------------------------------------------------------------------------
// Flash attention per (b, h, 128 q rows). Q pre-scaled by 1/TEMP.
// qh: [B,H,LQ,64] bf16   kh: [B,H,LK,64] bf16   vhT: [B,H,64,LK] bf16
// ctx: [B,LQ,H*64] bf16  Mrow/Lrow: [B,H,LQ] fp32 softmax stats
__global__ __launch_bounds__(256) void flash_attn(const bf16_t* __restrict__ qh,
                                                  const bf16_t* __restrict__ kh,
                                                  const bf16_t* __restrict__ vhT,
                                                  const unsigned char* __restrict__ mask,
                                                  bf16_t* __restrict__ ctx,
                                                  float* __restrict__ Mrow,
                                                  float* __restrict__ Lrow) {
    __shared__ bf16_t pshare[8 * 16 * 32];   // 8 KB, per-wave P staging
    const int lane = threadIdx.x & 31;
    const int w    = threadIdx.x >> 5;
    const int b    = blockIdx.z, h = blockIdx.y;
    const int q0   = blockIdx.x * 128 + w * 16;
    const int cl = lane & 15, half = lane >> 4;

    const size_t qoff = (((size_t)b * HHn + h) * LQn + q0) * 64;
    const v16bf aq0 = load_a16(qh + qoff, lane, 64);        // dk 0..31
    const v16bf aq1 = load_a16(qh + qoff + 32, lane, 64);   // dk 32..63

    float mrow[8], lrow[8];
    v8f o[4] = {};
#pragma unroll
    for (int r = 0; r < 8; r++) { mrow[r] = -1e30f; lrow[r] = 0.f; }

    bf16_t* myp = pshare + w * (16 * 32);
    const bf16_t* khbase = kh + ((size_t)b * HHn + h) * LKn * 64;

    for (int k0 = 0; k0 < LKn; k0 += 32) {
        if (k0 + 32 < LKn)
            __builtin_prefetch(khbase + (size_t)(k0 + 32) * 64, 0, 0);
        v8f s[2];
#pragma unroll
        for (int t = 0; t < 2; t++) {
            const int kt = k0 + t * 16;
            const bf16_t* kb = khbase + (size_t)kt * 64;
            v8f z = {};
            z = wmma_bf16(aq0, load_b16(kb, lane, 64), z);
            z = wmma_bf16(aq1, load_b16(kb + 32, lane, 64), z);
            if (mask[b * LKn + kt + cl]) {
#pragma unroll
                for (int r = 0; r < 8; r++) z[r] = -1e30f;
            }
            s[t] = z;
        }
        // online softmax: row = accumulator index (+8 for upper half-lanes),
        // columns live across the 16 lanes of the half -> shfl_xor reduction
        float alpha[8];
#pragma unroll
        for (int r = 0; r < 8; r++) {
            float v = fmaxf(s[0][r], s[1][r]);
            v = fmaxf(v, __shfl_xor(v, 1));
            v = fmaxf(v, __shfl_xor(v, 2));
            v = fmaxf(v, __shfl_xor(v, 4));
            v = fmaxf(v, __shfl_xor(v, 8));
            const float mnew = fmaxf(mrow[r], v);
            const float p0 = __expf(s[0][r] - mnew);
            const float p1 = __expf(s[1][r] - mnew);
            s[0][r] = p0; s[1][r] = p1;
            float ps = p0 + p1;
            ps += __shfl_xor(ps, 1);
            ps += __shfl_xor(ps, 2);
            ps += __shfl_xor(ps, 4);
            ps += __shfl_xor(ps, 8);
            alpha[r] = __expf(mrow[r] - mnew);
            lrow[r] = lrow[r] * alpha[r] + ps;
            mrow[r] = mnew;
        }
#pragma unroll
        for (int nt = 0; nt < 4; nt++)
#pragma unroll
            for (int r = 0; r < 8; r++) o[nt][r] *= alpha[r];

        // C-layout -> A-layout re-fragment through wave-private LDS (bf16)
#pragma unroll
        for (int t = 0; t < 2; t++)
#pragma unroll
            for (int r = 0; r < 8; r++)
                myp[(r + half * 8) * 32 + t * 16 + cl] = (bf16_t)s[t][r];
        asm volatile("s_wait_dscnt 0" ::: "memory");   // wave's DS ops are in-order
        const v16bf ap = load_a16(myp, lane, 32);
#pragma unroll
        for (int nt = 0; nt < 4; nt++) {
            const bf16_t* vb = vhT + (((size_t)b * HHn + h) * 64 + nt * 16) * LKn + k0;
            o[nt] = wmma_bf16(ap, load_b16(vb, lane, LKn), o[nt]);
        }
    }

    // normalize and write ctx [B, LQ, H*64]
#pragma unroll
    for (int nt = 0; nt < 4; nt++)
#pragma unroll
        for (int r = 0; r < 8; r++) {
            const int q = q0 + r + half * 8;
            const int d = nt * 16 + cl;
            ctx[((size_t)b * LQn + q) * HDn + h * 64 + d] = (bf16_t)(o[nt][r] / lrow[r]);
        }
    if (cl == 0) {
#pragma unroll
        for (int r = 0; r < 8; r++) {
            const int q = q0 + r + half * 8;
            Mrow[((size_t)b * HHn + h) * LQn + q] = mrow[r];
            Lrow[((size_t)b * HHn + h) * LQn + q] = lrow[r];
        }
    }
}

// ---------------------------------------------------------------------------
// attn_avg: recompute S per head with WMMA, normalize with stored (m, l),
// average over all heads in-register (no atomics), write fp32.
__global__ __launch_bounds__(256) void attn_avg(const bf16_t* __restrict__ qh,
                                                const bf16_t* __restrict__ kh,
                                                const unsigned char* __restrict__ mask,
                                                const float* __restrict__ Mrow,
                                                const float* __restrict__ Lrow,
                                                float* __restrict__ attn) {
    const int lane = threadIdx.x & 31;
    const int w    = threadIdx.x >> 5;
    const int bq   = blockIdx.x;            // 0 .. B*LQ/16-1
    const int b    = bq >> 7;               // 128 q-tiles per batch
    const int q0   = (bq & 127) * 16;
    const int k0   = blockIdx.y * 128 + w * 16;
    const int cl = lane & 15, half = lane >> 4;

    const int key = k0 + cl;
    const bool km = mask[b * LKn + key] != 0;
    v8f acc = {};

    for (int h = 0; h < HHn; h++) {
        const size_t qoff = (((size_t)b * HHn + h) * LQn + q0) * 64;
        const v16bf a0 = load_a16(qh + qoff, lane, 64);
        const v16bf a1 = load_a16(qh + qoff + 32, lane, 64);
        const bf16_t* kb = kh + (((size_t)b * HHn + h) * LKn + k0) * 64;
        v8f s = {};
        s = wmma_bf16(a0, load_b16(kb, lane, 64), s);
        s = wmma_bf16(a1, load_b16(kb + 32, lane, 64), s);
#pragma unroll
        for (int r = 0; r < 8; r++) {
            const int q = q0 + r + half * 8;
            const float m = Mrow[((size_t)b * HHn + h) * LQn + q];
            const float l = Lrow[((size_t)b * HHn + h) * LQn + q];
            acc[r] += km ? 0.f : __expf(s[r] - m) / l;
        }
    }
#pragma unroll
    for (int r = 0; r < 8; r++) {
        const int q = q0 + r + half * 8;
        attn[((size_t)b * LQn + q) * LKn + key] = acc[r] * (1.0f / HHn);
    }
}

// ---------------------------------------------------------------------------
// Final LayerNorm over D=1024 per row (residual already fused in out-proj)
__global__ __launch_bounds__(256) void layer_norm(const float* __restrict__ y,
                                                  const float* __restrict__ g,
                                                  const float* __restrict__ beta,
                                                  float* __restrict__ out) {
    __shared__ float red[2][8];
    const int row = blockIdx.x;
    const float* xr = y + (size_t)row * DDn;
    float s = 0.f, ss = 0.f;
    for (int i = threadIdx.x; i < DDn; i += 256) {
        const float v = xr[i];
        s += v; ss += v * v;
    }
#pragma unroll
    for (int off = 16; off > 0; off >>= 1) { s += __shfl_xor(s, off); ss += __shfl_xor(ss, off); }
    const int w = threadIdx.x >> 5, lane = threadIdx.x & 31;
    if (lane == 0) { red[0][w] = s; red[1][w] = ss; }
    __syncthreads();
    if (w == 0) {
        s  = (lane < 8) ? red[0][lane] : 0.f;
        ss = (lane < 8) ? red[1][lane] : 0.f;
#pragma unroll
        for (int off = 4; off > 0; off >>= 1) { s += __shfl_xor(s, off); ss += __shfl_xor(ss, off); }
        if (lane == 0) { red[0][0] = s; red[1][0] = ss; }
    }
    __syncthreads();
    const float mean = red[0][0] * (1.0f / DDn);
    const float var  = red[1][0] * (1.0f / DDn) - mean * mean;
    const float rstd = rsqrtf(var + LN_EPS);
    for (int i = threadIdx.x; i < DDn; i += 256)
        out[(size_t)row * DDn + i] = (xr[i] - mean) * rstd * g[i] + beta[i];
}

// ---------------------------------------------------------------------------
extern "C" void kernel_launch(void* const* d_in, const int* in_sizes, int n_in,
                              void* d_out, int out_size, void* d_ws, size_t ws_size,
                              hipStream_t stream) {
    const float* q   = (const float*)d_in[0];
    const float* k   = (const float*)d_in[1];
    const float* v   = (const float*)d_in[2];
    const unsigned char* mask = (const unsigned char*)d_in[3];   // bool array
    const float* Wq  = (const float*)d_in[4];
    const float* bq  = (const float*)d_in[5];
    const float* Wk  = (const float*)d_in[6];
    const float* bk  = (const float*)d_in[7];
    const float* Wv  = (const float*)d_in[8];
    const float* bv  = (const float*)d_in[9];
    const float* Wo  = (const float*)d_in[10];
    const float* bo  = (const float*)d_in[11];
    const float* lng = (const float*)d_in[12];
    const float* lnb = (const float*)d_in[13];
    (void)in_sizes; (void)n_in; (void)out_size; (void)ws_size;

    const size_t M = (size_t)BB * LQn;          // 8192 token rows

    char* ws = (char*)d_ws;
    size_t off = 0;
    auto alloc = [&](size_t bytes) -> void* {
        void* p = ws + off;
        off += (bytes + 255) & ~(size_t)255;
        return p;
    };
    bf16_t* qb  = (bf16_t*)alloc(M * DDn * 2);
    bf16_t* kb  = (bf16_t*)alloc(M * DDn * 2);
    bf16_t* vb  = (bf16_t*)alloc(M * DDn * 2);
    bf16_t* WqT = (bf16_t*)alloc((size_t)DDn * DDn * 2);
    bf16_t* WkT = (bf16_t*)alloc((size_t)DDn * DDn * 2);
    bf16_t* WvT = (bf16_t*)alloc((size_t)DDn * DDn * 2);
    bf16_t* WoT = (bf16_t*)alloc((size_t)DDn * DDn * 2);
    bf16_t* qhB = (bf16_t*)alloc(M * 64 * HHn * 2);   // [B,H,LQ,64]
    bf16_t* khB = (bf16_t*)alloc(M * 64 * HHn * 2);   // [B,H,LK,64]
    bf16_t* vhT = (bf16_t*)alloc(M * 64 * HHn * 2);   // [B,H,64,LK]
    bf16_t* ctx = (bf16_t*)alloc(M * HDn * 2);        // [B,LQ,H*64]
    float*  yws = (float*)alloc(M * DDn * 4);         // out-proj + residual
    float*  Mrw = (float*)alloc((size_t)BB * HHn * LQn * 4);
    float*  Lrw = (float*)alloc((size_t)BB * HHn * LQn * 4);

    const int nTok = (int)(M * DDn);
    cvt_bf16<<<2048, 256, 0, stream>>>(q, qb, nTok);
    cvt_bf16<<<2048, 256, 0, stream>>>(k, kb, nTok);
    cvt_bf16<<<2048, 256, 0, stream>>>(v, vb, nTok);
    pack_wT<<<1024, 256, 0, stream>>>(Wq, WqT, DDn, DDn);
    pack_wT<<<1024, 256, 0, stream>>>(Wk, WkT, DDn, DDn);
    pack_wT<<<1024, 256, 0, stream>>>(Wv, WvT, DDn, DDn);
    pack_wT<<<1024, 256, 0, stream>>>(Wo, WoT, DDn, DDn);

    dim3 gGemm((unsigned)(M / 256), DDn / 64);
    // Q projection, pre-scaled by 1/sqrt(dk); head-major output
    gemm_bf16<0><<<gGemm, 256, 0, stream>>>(qb, WqT, bq, nullptr, qhB,
                                            (int)M, DDn, DDn, INV_TEMP);
    gemm_bf16<0><<<gGemm, 256, 0, stream>>>(kb, WkT, bk, nullptr, khB,
                                            (int)M, DDn, DDn, 1.0f);
    gemm_bf16<1><<<gGemm, 256, 0, stream>>>(vb, WvT, bv, nullptr, vhT,
                                            (int)M, DDn, DDn, 1.0f);

    dim3 gFlash(LQn / 128, HHn, BB);
    flash_attn<<<gFlash, 256, 0, stream>>>(qhB, khB, vhT, mask, ctx, Mrw, Lrw);

    float* attn_out = (float*)d_out + M * DDn;        // second tuple element
    dim3 gAvg((unsigned)(M / 16), LKn / 128);
    attn_avg<<<gAvg, 256, 0, stream>>>(qhB, khB, mask, Mrw, Lrw, attn_out);

    // out projection + residual (fp32), then LayerNorm into d_out
    gemm_bf16<2><<<gGemm, 256, 0, stream>>>(ctx, WoT, bo, q, yws,
                                            (int)M, DDn, DDn, 1.0f);
    layer_norm<<<(unsigned)M, 256, 0, stream>>>(yws, lng, lnb, (float*)d_out);
}
